// ReadoutLayer_8246337208358
// MI455X (gfx1250) — matrix operand order repaired and checked by reading, not compiled, not run on hardware
//
#include <hip/hip_runtime.h>
#include <hip/hip_fp16.h>

typedef __attribute__((ext_vector_type(16))) _Float16 v16h;
typedef __attribute__((ext_vector_type(8)))  float    v8f;

#define B_    64
#define T_    1024
#define C_    1024
#define D_    256
#define DT    64            // D columns per block (4 WMMA n-tiles per wave)
#define NT    (DT / 16)
#define TSEG  4             // T segments (one per wave in the block)
#define TLEN  (T_ / TSEG)   // 256
#define HALO  64            // beta^64 ~ 1.2e-10: warm-up exact at fp32

// ---------------- prep: w -> f16 wT [D][C], norm[d] = sum_c w^2 ----------------
__global__ void readout_prep(const float* __restrict__ w,
                             __half* __restrict__ wT,
                             float* __restrict__ norm) {
  int d = blockIdx.x * blockDim.x + threadIdx.x;
  if (d >= D_) return;
  float acc = 0.f;
  for (int c = 0; c < C_; ++c) {
    float v = w[(size_t)c * D_ + d];
    acc += v * v;
    wT[(size_t)d * C_ + c] = __float2half(v);
  }
  norm[d] = acc;
}

__global__ void readout_init_pmax(unsigned int* __restrict__ pmax) {
  int i = blockIdx.x * blockDim.x + threadIdx.x;
  if (i < B_ * D_) pmax[i] = 0u;   // below sortable encoding of any real float
}

// ---------------- main fused GEMM + scan + max ----------------
__global__ void __launch_bounds__(128)
readout_main(const float* __restrict__ x, const __half* __restrict__ wT,
             const float* __restrict__ beta_p, unsigned int* __restrict__ pmax) {
  extern __shared__ __half smemB[];          // [DT][C_] f16 = 128 KB
  const int b     = blockIdx.x;
  const int dbase = blockIdx.y * DT;
  const int lane  = threadIdx.x & 31;
  const int wv    = threadIdx.x >> 5;        // T-segment id, 0..3
  const int lhalf = lane & 15;
  const int hsel  = lane >> 4;               // 0: rows 0-7 / k-lo, 1: rows 8-15 / k-hi

  // Stage w^T tile into LDS once per block (reused by all chunks & waves).
  {
    const uint4* src = (const uint4*)(wT + (size_t)dbase * C_);
    uint4* dst = (uint4*)smemB;
    for (int i = threadIdx.x; i < DT * C_ / 8; i += blockDim.x) dst[i] = src[i];
  }
  __syncthreads();

  const float beta = beta_p[0];
  const float omb  = 1.0f - beta;
  float bp[8];                               // bp[i] = beta^(i+1)
  {
    float p = beta;
#pragma unroll
    for (int i = 0; i < 8; ++i) { bp[i] = p; p *= beta; }
  }
  const float bp8 = bp[7];                   // beta^8

  const int t_emit  = wv * TLEN;
  const int t_begin = (wv == 0) ? 0 : (t_emit - HALO);
  const int t_end   = t_emit + TLEN;

  float carry[NT], runmax[NT];
#pragma unroll
  for (int n = 0; n < NT; ++n) { carry[n] = 0.f; runmax[n] = -3.402823466e38f; }

  for (int t0 = t_begin; t0 < t_end; t0 += 16) {
    v8f acc0[NT], accL[NT];
#pragma unroll
    for (int n = 0; n < NT; ++n) { v8f z = {}; acc0[n] = z; accL[n] = z; }

    const float* xrow = x + ((size_t)b * T_ + (size_t)(t0 + lhalf)) * C_;

#pragma unroll 2
    for (int c0 = 0; c0 < C_; c0 += 32) {
      // ---- A fragment (16x32 f16 layout), split x into hi + 2^-11 * lo ----
      const float* pa = xrow + c0 + hsel * 16;
      float4 f0 = *(const float4*)(pa + 0);
      float4 f1 = *(const float4*)(pa + 4);
      float4 f2 = *(const float4*)(pa + 8);
      float4 f3 = *(const float4*)(pa + 12);
      float fv[16] = {f0.x, f0.y, f0.z, f0.w, f1.x, f1.y, f1.z, f1.w,
                      f2.x, f2.y, f2.z, f2.w, f3.x, f3.y, f3.z, f3.w};
      v16h a_hi, a_lo;
#pragma unroll
      for (int j = 0; j < 16; ++j) {
        _Float16 h = (_Float16)fv[j];
        a_hi[j] = h;
        a_lo[j] = (_Float16)((fv[j] - (float)h) * 2048.0f);
      }
      // ---- load ALL B fragments first (one dscnt wait covers all 8 ds loads) ----
      v16h bfrag[NT];
#pragma unroll
      for (int n = 0; n < NT; ++n) {
        const __half* pb = smemB + ((n * 16 + lhalf) * C_ + c0 + hsel * 16);
        bfrag[n] = *(const v16h*)pb;
      }
      // ---- WMMAs grouped by A operand: 4x a_hi, then 4x a_lo ----
#pragma unroll
      for (int n = 0; n < NT; ++n)
        acc0[n] = __builtin_amdgcn_wmma_f32_16x16x32_f16(
            false, a_hi, false, bfrag[n], (short)0, acc0[n], false, false);
#pragma unroll
      for (int n = 0; n < NT; ++n)
        accL[n] = __builtin_amdgcn_wmma_f32_16x16x32_f16(
            false, a_lo, false, bfrag[n], (short)0, accL[n], false, false);
    }

    // ---- in-register leaky-integrator scan across the 16 rows + running max ----
    const bool emit = (t0 >= t_emit);
#pragma unroll
    for (int n = 0; n < NT; ++n) {
      float rs[8];
      float s = 0.f;
#pragma unroll
      for (int i = 0; i < 8; ++i) {
        float h = acc0[n][i] + accL[n][i] * (1.0f / 2048.0f);
        s = fmaf(beta, s, omb * h);          // local scan, zero init
        rs[i] = s;
      }
      float s7low = __shfl(s, lhalf, 32);    // lower half's local mem after row 7
      float ch = hsel ? fmaf(bp8, carry[n], s7low) : carry[n];
#pragma unroll
      for (int i = 0; i < 8; ++i) rs[i] = fmaf(bp[i], ch, rs[i]);  // full mem_t
      carry[n] = __shfl(rs[7], 16 + lhalf, 32);                    // mem at row 15
      if (emit) {
#pragma unroll
        for (int i = 0; i < 8; ++i) runmax[n] = fmaxf(runmax[n], rs[i]);
      }
    }
  }

  // ---- merge lane halves, then merge T-segments via sortable-uint atomic max ----
#pragma unroll
  for (int n = 0; n < NT; ++n) {
    float m2 = fmaxf(runmax[n], __shfl_xor(runmax[n], 16, 32));
    if (lane < 16) {
      unsigned u = __float_as_uint(m2);
      u = (u & 0x80000000u) ? ~u : (u | 0x80000000u);   // monotone encoding
      atomicMax(&pmax[b * D_ + dbase + n * 16 + lane], u);
    }
  }
}

// ---------------- finalize: out = max / (norm + eps) - bias ----------------
__global__ void readout_finalize(const unsigned int* __restrict__ pmax,
                                 const float* __restrict__ norm,
                                 const float* __restrict__ bias,
                                 float* __restrict__ out) {
  int i = blockIdx.x * blockDim.x + threadIdx.x;
  if (i >= B_ * D_) return;
  unsigned u = pmax[i];
  unsigned bits = (u & 0x80000000u) ? (u & 0x7fffffffu) : ~u;
  float m = __uint_as_float(bits);
  int d = i & (D_ - 1);
  out[i] = m / (norm[d] + 1e-8f) - bias[d];
}

extern "C" void kernel_launch(void* const* d_in, const int* in_sizes, int n_in,
                              void* d_out, int out_size, void* d_ws, size_t ws_size,
                              hipStream_t stream) {
  const float* x    = (const float*)d_in[0];
  const float* w    = (const float*)d_in[1];
  const float* bias = (const float*)d_in[2];
  const float* beta = (const float*)d_in[3];
  float* out = (float*)d_out;

  char* ws = (char*)d_ws;
  __half*       wT   = (__half*)ws;                          // 512 KB
  float*        norm = (float*)(ws + 512 * 1024);            // 1 KB
  unsigned int* pmax = (unsigned int*)(ws + 512 * 1024 + 4096);  // 64 KB

  const int smem_bytes = DT * C_ * (int)sizeof(__half);      // 128 KB
  (void)hipFuncSetAttribute((const void*)readout_main,
                            hipFuncAttributeMaxDynamicSharedMemorySize, smem_bytes);

  readout_prep<<<1, 256, 0, stream>>>(w, wT, norm);
  readout_init_pmax<<<(B_ * D_) / 256, 256, 0, stream>>>(pmax);

  dim3 grid(B_, D_ / DT);                                    // 64 x 4 blocks
  readout_main<<<grid, 128, smem_bytes, stream>>>(x, wT, beta, pmax);

  readout_finalize<<<(B_ * D_) / 256, 256, 0, stream>>>(pmax, norm, bias, out);
}
